// QuantPINN_NLSE_901943132480
// MI455X (gfx1250) — 8-bit matrix operand order repaired and checked by reading, compiled, not run on hardware
//
#include <hip/hip_runtime.h>
#include <cmath>


typedef __attribute__((ext_vector_type(8))) int v8i;
typedef __attribute__((ext_vector_type(4))) int v4i;
typedef __attribute__((address_space(1))) v4i* gv4i_p;
typedef __attribute__((address_space(3))) v4i* lv4i_p;

#define HID   100
#define KPAD  128
#define NPAD  112
#define ROWS_PER_BLOCK 128
#define QMAXF 127.0f
#define EPSS  1e-12f

#define WS_ACT   0
#define WS_SW    32
#define WS_BQ    64
#define WS_H     102400
#define BQ_BYTES (NPAD * KPAD)

struct WPtrs { const float* W[7]; };

__device__ __forceinline__ float act_scale(const unsigned* actmax, int slot) {
    float m = __uint_as_float(actmax[slot]);
    return fmaxf(m * (1.0f / QMAXF), EPSS);
}

__device__ __forceinline__ int quant127(float x, float inv_s) {
    float r = rintf(x * inv_s);
    r = fminf(fmaxf(r, -QMAXF), QMAXF);
    return (int)r;
}

__device__ __forceinline__ void wave_max_atomic(float m, unsigned* slot) {
#pragma unroll
    for (int i = 16; i >= 1; i >>= 1) m = fmaxf(m, __shfl_xor(m, i, 32));
    if ((threadIdx.x & 31) == 0) atomicMax(slot, __float_as_uint(m));
}

__global__ void init_slots_kernel(unsigned* actmax) {
    if (threadIdx.x < 8) actmax[threadIdx.x] = 0u;
}

__global__ void wquant_kernel(WPtrs p, float* swArr, signed char* Bq) {
    const int L    = blockIdx.x;
    const int rows = (L == 6) ? 2 : HID;
    const int cols = (L == 0) ? 2 : HID;
    const float* W = p.W[L];
    const int tid  = threadIdx.x;

    __shared__ float red[256];
    __shared__ float s_sw;
    float m = 0.0f;
    for (int i = tid; i < rows * cols; i += 256) m = fmaxf(m, fabsf(W[i]));
    red[tid] = m;
    __syncthreads();
    for (int s = 128; s > 0; s >>= 1) {
        if (tid < s) red[tid] = fmaxf(red[tid], red[tid + s]);
        __syncthreads();
    }
    if (tid == 0) {
        float sw = fmaxf(red[0] * (1.0f / QMAXF), EPSS);
        swArr[L] = sw;
        s_sw     = sw;
    }
    __syncthreads();
    const float inv_sw = 1.0f / s_sw;

    signed char* dst = Bq + (size_t)L * BQ_BYTES;
    for (int i = tid; i < BQ_BYTES; i += 256) {
        int n = i >> 7;
        int k = i & (KPAD - 1);
        float v = (n < rows && k < cols) ? W[n * cols + k] : 0.0f;
        dst[i] = (signed char)quant127(v, inv_sw);
    }
}

__global__ void maxabs_kernel(const float* __restrict__ x, int n, unsigned* slot) {
    float m = 0.0f;
    for (int i = blockIdx.x * blockDim.x + threadIdx.x; i < n; i += gridDim.x * blockDim.x)
        m = fmaxf(m, fabsf(x[i]));
    wave_max_atomic(m, slot);
}

__global__ void layer1_kernel(const float* __restrict__ z, const float* __restrict__ t,
                              const signed char* __restrict__ Bq, const float* __restrict__ b1,
                              const unsigned* __restrict__ actmax, const float* __restrict__ swArr,
                              float* __restrict__ h, int n, unsigned* nextSlot) {
    int idx = blockIdx.x * blockDim.x + threadIdx.x;
    float v = 0.0f;
    if (idx < n * HID) {
        int row = idx / HID;
        int col = idx - row * HID;

        float sx = act_scale(actmax, 0);
        float sw = swArr[0];
        float inv_sx = 1.0f / sx;
        int xi0 = quant127(z[row], inv_sx);
        int xi1 = quant127(t[row], inv_sx);
        int w0  = (int)Bq[col * KPAD + 0];
        int w1  = (int)Bq[col * KPAD + 1];
        int acc = xi0 * w0 + xi1 * w1;

        float sb = sx * sw;
        float bi = fminf(fmaxf(rintf(b1[col] / sb), -128.0f), 127.0f);
        v = tanhf(sb * ((float)acc + bi));
        h[idx] = v;
    }
    wave_max_atomic(fabsf(v), nextSlot);
}

template <int NOUT, int NTILES, bool TANH, bool REDUCE, bool FULL>
__global__ void qgemm_kernel(const float* __restrict__ hin, const signed char* __restrict__ Bq,
                             const float* __restrict__ bias, const unsigned* __restrict__ actmax,
                             int actSlot, const float* __restrict__ swArr, int swSlot,
                             float* __restrict__ out, int ntotal, unsigned* nextSlot) {
    __shared__ signed char Alds[ROWS_PER_BLOCK * KPAD];
    __shared__ signed char Blds[NPAD * KPAD];
    __shared__ float       biasLds[NPAD];

    const int tid = threadIdx.x;
    const long long rowbase = (long long)blockIdx.x * ROWS_PER_BLOCK;

    const float sx = act_scale(actmax, actSlot);
    const float sw = swArr[swSlot];
    const float sb = sx * sw;
    const float inv_sx = 1.0f / sx;

#if __has_builtin(__builtin_amdgcn_global_load_async_to_lds_b128)
    for (int i = tid; i < (NPAD * KPAD) / 16; i += 256) {
        __builtin_amdgcn_global_load_async_to_lds_b128(
            (gv4i_p)(Bq + i * 16),
            (lv4i_p)(Blds + i * 16),
             0,  0);
    }
#else
    {
        const int* src = (const int*)Bq;
        int* dst = (int*)Blds;
        for (int i = tid; i < (NPAD * KPAD) / 4; i += 256) dst[i] = src[i];
    }
#endif

    if (tid < NPAD) {
        float bi = 0.0f;
        if (tid < NOUT) bi = fminf(fmaxf(rintf(bias[tid] / sb), -128.0f), 127.0f);
        biasLds[tid] = bi;
    }

    {
        int* dst = (int*)Alds;
        for (int i = tid; i < (ROWS_PER_BLOCK * KPAD) / 4; i += 256) {
            int row = i >> 5;
            int k0  = (i & 31) * 4;
            long long grow = rowbase + row;
            unsigned packed = 0u;
            if ((FULL || grow < ntotal) && k0 < HID) {
                const float4 v4 = *(const float4*)(hin + grow * HID + k0);
                packed  = (unsigned)(quant127(v4.x, inv_sx) & 0xFF);
                packed |= (unsigned)(quant127(v4.y, inv_sx) & 0xFF) << 8;
                packed |= (unsigned)(quant127(v4.z, inv_sx) & 0xFF) << 16;
                packed |= (unsigned)(quant127(v4.w, inv_sx) & 0xFF) << 24;
            }
            dst[i] = (int)packed;
        }
    }

#if __has_builtin(__builtin_amdgcn_global_load_async_to_lds_b128)
#if __has_builtin(__builtin_amdgcn_s_wait_asynccnt)
    __builtin_amdgcn_s_wait_asynccnt(0);
#else
    asm volatile("s_wait_asynccnt 0x0" ::: "memory");
#endif
#endif
    __syncthreads();

    const int wave = tid >> 5;
    const int lane = tid & 31;
    const int half = lane >> 4;
    const int lr   = lane & 15;

    const signed char* Arow = &Alds[(wave * 16 + lr) * KPAD];
    v8i a0, a1;
#pragma unroll
    for (int d = 0; d < 8; ++d) {
        int koffA = ((d >> 2) & 1) * 32 + ((d >> 1) & 1) * 16 + (d & 1) * 4 + half * 8;
        a0[d] = *(const int*)&Arow[koffA];
        a1[d] = *(const int*)&Arow[64 + koffA];
    }

    float wmax = 0.0f;
    for (int ntile = 0; ntile < NTILES; ++ntile) {
        const signed char* Brow = &Blds[(ntile * 16 + lr) * KPAD];
        v8i b0, b1;
#pragma unroll
        for (int d = 0; d < 8; ++d) {
            int koffB = ((d >> 2) & 1) * 32 + ((d >> 1) & 1) * 16 + (d & 1) * 4 + half * 8;
            b0[d] = *(const int*)&Brow[koffB];
            b1[d] = *(const int*)&Brow[64 + koffB];
        }
        v8i acc = {};
        acc = __builtin_amdgcn_wmma_i32_16x16x64_iu8(true, a0, true, b0, acc, false, false);
        acc = __builtin_amdgcn_wmma_i32_16x16x64_iu8(true, a1, true, b1, acc, false, false);

        const int ncol = ntile * 16 + lr;
        if (ncol < NOUT) {
            float bi = biasLds[ncol];
#pragma unroll
            for (int j = 0; j < 8; ++j) {
                long long row = rowbase + wave * 16 + half * 8 + j;
                if (FULL || row < ntotal) {
                    float v = sb * ((float)acc[j] + bi);
                    if (TANH) v = tanhf(v);
                    out[row * NOUT + ncol] = v;
                    if (REDUCE) wmax = fmaxf(wmax, fabsf(v));
                }
            }
        }
    }
    if (REDUCE) wave_max_atomic(wmax, nextSlot);
}

extern "C" void kernel_launch(void* const* d_in, const int* in_sizes, int n_in,
                              void* d_out, int out_size, void* d_ws, size_t ws_size,
                              hipStream_t stream) {
    (void)n_in; (void)out_size; (void)ws_size;

    const float* z = (const float*)d_in[0];
    const float* t = (const float*)d_in[1];
    const int N = in_sizes[0];

    WPtrs wp;
    const float* bptr[7];
    for (int L = 0; L < 7; ++L) {
        wp.W[L] = (const float*)d_in[2 + 2 * L];
        bptr[L] = (const float*)d_in[3 + 2 * L];
    }

    char* ws = (char*)d_ws;
    unsigned*    actmax = (unsigned*)(ws + WS_ACT);
    float*       swArr  = (float*)(ws + WS_SW);
    signed char* Bq     = (signed char*)(ws + WS_BQ);
    float*       h      = (float*)(ws + WS_H);

    const int  gemmBlocks = (N + ROWS_PER_BLOCK - 1) / ROWS_PER_BLOCK;
    const bool full       = (N % ROWS_PER_BLOCK) == 0;

    init_slots_kernel<<<1, 32, 0, stream>>>(actmax);
    wquant_kernel<<<7, 256, 0, stream>>>(wp, swArr, Bq);

    maxabs_kernel<<<1024, 256, 0, stream>>>(z, N, actmax + 0);
    maxabs_kernel<<<1024, 256, 0, stream>>>(t, N, actmax + 0);
    layer1_kernel<<<(N * HID + 255) / 256, 256, 0, stream>>>(
        z, t, Bq, bptr[0], actmax, swArr, h, N, actmax + 1);

    for (int L = 1; L <= 5; ++L) {
        if (full)
            qgemm_kernel<HID, 7, true, true, true><<<gemmBlocks, 256, 0, stream>>>(
                h, Bq + (size_t)L * BQ_BYTES, bptr[L], actmax, L, swArr, L, h, N, actmax + L + 1);
        else
            qgemm_kernel<HID, 7, true, true, false><<<gemmBlocks, 256, 0, stream>>>(
                h, Bq + (size_t)L * BQ_BYTES, bptr[L], actmax, L, swArr, L, h, N, actmax + L + 1);
    }

    if (full)
        qgemm_kernel<2, 1, false, false, true><<<gemmBlocks, 256, 0, stream>>>(
            h, Bq + (size_t)6 * BQ_BYTES, bptr[6], actmax, 6, swArr, 6, (float*)d_out, N, nullptr);
    else
        qgemm_kernel<2, 1, false, false, false><<<gemmBlocks, 256, 0, stream>>>(
            h, Bq + (size_t)6 * BQ_BYTES, bptr[6], actmax, 6, swArr, 6, (float*)d_out, N, nullptr);
}
